// LayerBlock_82575041232935
// MI455X (gfx1250) — matrix operand order repaired and checked by reading, not compiled, not run on hardware
//
#include <hip/hip_runtime.h>
#include <hip/hip_bf16.h>

typedef __attribute__((ext_vector_type(16))) __bf16 v16bf;
typedef __attribute__((ext_vector_type(8)))  float  v8f;

#define E_TOT 40000
#define LC    512
#define H_DIM 256
#define EC    256

// ---------------- workspace fragment layout ----------------
// One B-fragment = 32x16 bf16 (hi) + 32x16 bf16 (lo) = 1024 halves = 2 KB.
// Within a fragment: lane L owns halves [L*32, L*32+16) = hi, [L*32+16, L*32+32) = lo.
// Element (L, h) of hi/lo = src_rows[(nt*16 + (L&15))*rs + ks*32 + ((L>=16)?8:0) + (h<8 ? h : h+8)]
// Regions (in 2KB fragment units):
//   [0,    1024): edge W,  [rowblock(8)=m*2+i][ks(8)][nt(16)]
//   [1024, 3072): w1,      [im(8)=i*4+m][ks(16)][nt(16)]
//   [3072, 5120): w2,      [im(8)=i*4+m][ks(8)][nt(32)]
#define FRAG_HALVES 1024
#define NFRAG_TOTAL 5120

// ---------------- pre-pass: split f32 weights to bf16 hi/lo, swizzled ----------------
__global__ void __launch_bounds__(256)
prep_weights(const float* __restrict__ Wd,   // (2048, 256)
             const float* __restrict__ w1,   // (2, 4, 256, 512)
             const float* __restrict__ w2,   // (2, 4, 512, 256)
             __bf16* __restrict__ ws) {
  const int lane = threadIdx.x & 31;
  const int f    = blockIdx.x * 8 + (threadIdx.x >> 5);   // fragment id
  if (f >= NFRAG_TOTAL) return;

  const float* src;
  int rs, ks, nt;
  if (f < 1024) {                       // edge weights
    const int rb  = f >> 7;
    const int rem = f & 127;
    ks = rem >> 4;  nt = rem & 15;
    src = Wd + (size_t)rb * (H_DIM * EC);
    rs  = EC;
  } else if (f < 3072) {                // w1
    const int g   = f - 1024;
    const int im  = g >> 8;
    const int rem = g & 255;
    ks = rem >> 4;  nt = rem & 15;
    src = w1 + (size_t)im * (H_DIM * LC);
    rs  = LC;
  } else {                              // w2
    const int g   = f - 3072;
    const int im  = g >> 8;
    const int rem = g & 255;
    ks = rem >> 5;  nt = rem & 31;
    src = w2 + (size_t)im * (LC * H_DIM);
    rs  = H_DIM;
  }

  const int n  = lane & 15;
  const int kb = (lane >> 4) << 3;      // 0 or 8
  const float* p = src + (size_t)(nt * 16 + n) * rs + ks * 32 + kb;

  v16bf hi, lo;
  #pragma unroll
  for (int h = 0; h < 16; ++h) {
    float fv = p[h < 8 ? h : h + 8];
    __bf16 hv = (__bf16)fv;
    hi[h] = hv;
    lo[h] = (__bf16)(fv - (float)hv);
  }
  __bf16* dst = ws + (size_t)f * FRAG_HALVES + lane * 32;
  *(v16bf*)dst        = hi;
  *(v16bf*)(dst + 16) = lo;
}

// ---------------- main kernel helpers ----------------
// Raw A fragment data (16 f32 per lane), ISA 7.12.2 A layout order:
// r[0..7] = k = b..b+7 ; r[8..15] = k = 16+b..23+b  (b = (L>=16)?8:0)
__device__ __forceinline__ void load_a_raw(const float* __restrict__ base, int rs, int lane,
                                           float r[16]) {
  const int rr = lane & 15;
  const int b  = (lane >> 4) << 3;
  const float* p = base + (size_t)rr * rs + b;      // 32-B aligned
  float4 a0 = ((const float4*)p)[0];
  float4 a1 = ((const float4*)p)[1];
  float4 c0 = ((const float4*)(p + 16))[0];
  float4 c1 = ((const float4*)(p + 16))[1];
  r[0] = a0.x;  r[1] = a0.y;  r[2]  = a0.z;  r[3]  = a0.w;
  r[4] = a1.x;  r[5] = a1.y;  r[6]  = a1.z;  r[7]  = a1.w;
  r[8] = c0.x;  r[9] = c0.y;  r[10] = c0.z;  r[11] = c0.w;
  r[12] = c1.x; r[13] = c1.y; r[14] = c1.z;  r[15] = c1.w;
}

__device__ __forceinline__ void cvt_a(const float r[16], v16bf& hi, v16bf& lo) {
  #pragma unroll
  for (int h = 0; h < 16; ++h) {
    __bf16 hv = (__bf16)r[h];
    hi[h] = hv;
    lo[h] = (__bf16)(r[h] - (float)hv);
  }
}

// fp32-accurate bf16x3: hi*hi + lo*hi + hi*lo, fp32 accumulate
__device__ __forceinline__ v8f wmma3(v16bf ah, v16bf al, v16bf bh, v16bf bl, v8f acc) {
  acc = __builtin_amdgcn_wmma_f32_16x16x32_bf16(false, ah, false, bh, (short)0, acc, false, false);
  acc = __builtin_amdgcn_wmma_f32_16x16x32_bf16(false, al, false, bh, (short)0, acc, false, false);
  acc = __builtin_amdgcn_wmma_f32_16x16x32_bf16(false, ah, false, bl, (short)0, acc, false, false);
  return acc;
}

// C(16x256) += A(16 x 32*KSTEPS) @ B ; B from pre-swizzled fragments [ks][NTTOT] (nt0 offset).
// A raw-prefetched one k-step ahead; B double-buffered one fragment ahead.
template<int KSTEPS, int NTTOT>
__device__ __forceinline__ void gemm16(const float* __restrict__ abase, int ars,
                                       const __bf16* __restrict__ fr, int nt0,
                                       int lane, v8f acc[16]) {
  const __bf16* lanep = fr + (size_t)nt0 * FRAG_HALVES + lane * 32;
  float ar[16];
  load_a_raw(abase, ars, lane, ar);

  #pragma unroll 1
  for (int ks = 0; ks < KSTEPS; ++ks) {
    v16bf ah, al;
    cvt_a(ar, ah, al);
    if (ks + 1 < KSTEPS) {
      load_a_raw(abase + (ks + 1) * 32, ars, lane, ar);   // prefetch next k-step (raw)
      __builtin_prefetch(lanep + (size_t)(ks + 1) * NTTOT * FRAG_HALVES, 0, 0);
    }

    const __bf16* fb = lanep + (size_t)ks * NTTOT * FRAG_HALVES;
    v16bf bh = *(const v16bf*)fb;
    v16bf bl = *(const v16bf*)(fb + 16);
    #pragma unroll
    for (int nt = 0; nt < 16; ++nt) {
      v16bf nbh = bh, nbl = bl;
      if (nt + 1 < 16) {
        const __bf16* fn = fb + (size_t)(nt + 1) * FRAG_HALVES;
        nbh = *(const v16bf*)fn;
        nbl = *(const v16bf*)(fn + 16);
      }
      acc[nt] = wmma3(ah, al, bh, bl, acc[nt]);   // overlaps with nbh/nbl loads
      bh = nbh; bl = nbl;
    }
  }
}

extern __shared__ float smem[];   // [0,32768): xe slices; [32768,65536): per-wave y staging

__global__ void __launch_bounds__(256)
fused_layerblock(const float* __restrict__ x,       // (E, 4, 2, 512)
                 const float* __restrict__ x_edge,  // (E, 256)
                 const float* __restrict__ bias,    // (2048)
                 const __bf16* __restrict__ wf,     // pre-swizzled weight fragments
                 float* __restrict__ out) {         // (E, 4, 2, 512)
  const int lane = threadIdx.x & 31;
  const int w    = threadIdx.x >> 5;                // 8 waves
  const int e0   = blockIdx.x * 16;

  float* xe_lds = smem;                             // xe[(i*4+m)*16 + e][256] f32
  float* ybuf   = smem + 32768 + w * (16 * H_DIM);  // this wave's y staging (16x256)

  const int col = lane & 15;                        // C/D layout: column
  const int rhi = (lane >> 4) << 3;                 // C/D layout: +8 rows for lanes 16..31

  // ---------------- Stage A: edge MLP, wave w owns xe slice (i,m) = (w&1, w>>1)
  {
    const int i_s = w & 1, m_s = w >> 1;
    const int rb  = m_s * 2 + i_s;                  // row-block in edge weight
    v8f acc[16];
    #pragma unroll
    for (int t = 0; t < 16; ++t)
      #pragma unroll
      for (int j = 0; j < 8; ++j) acc[t][j] = 0.0f;

    gemm16<EC / 32, 16>(x_edge + (size_t)e0 * EC, EC,
                        wf + (size_t)(rb * 128) * FRAG_HALVES, 0, lane, acc);

    const int slice = i_s * 4 + m_s;
    #pragma unroll
    for (int nt = 0; nt < 16; ++nt) {
      const int h  = nt * 16 + col;
      const float bv = bias[rb * H_DIM + h];
      #pragma unroll
      for (int j = 0; j < 8; ++j) {
        float a = acc[nt][j] + bv;
        float s = a / (1.0f + __expf(-a));          // silu
        xe_lds[(size_t)(slice * 16 + rhi + j) * H_DIM + h] = s;
      }
    }
  }
  __syncthreads();

  // ---------------- Main pipelines: wave w -> output row (m, p)
  // p=0 (pos): z(i=0,s=0) - z(i=1,s=1);  p=1 (neg): z(i=0,s=1) + z(i=1,s=0)
  const int m = w >> 1;
  const int p = w & 1;

  #pragma unroll 1
  for (int pipe = 0; pipe < 2; ++pipe) {
    const int   i    = pipe;
    const int   s    = (pipe == 0) ? p : 1 - p;
    const float sign = (pipe == 1 && p == 0) ? -1.0f : 1.0f;
    const int   im   = i * 4 + m;

    // GEMM1: y(16x256) = xr(16x512) @ w1[i][m]^T ; gate by xe ; stage to LDS
    {
      v8f acc[16];
      #pragma unroll
      for (int t = 0; t < 16; ++t)
        #pragma unroll
        for (int j = 0; j < 8; ++j) acc[t][j] = 0.0f;

      const float* abase = x + (size_t)e0 * 4096 + m * 1024 + s * 512;   // e-stride 4096
      gemm16<LC / 32, 16>(abase, 4096,
                          wf + (size_t)(1024 + im * 256) * FRAG_HALVES, 0, lane, acc);

      #pragma unroll
      for (int nt = 0; nt < 16; ++nt) {
        const int h = nt * 16 + col;
        #pragma unroll
        for (int j = 0; j < 8; ++j) {
          const int er = rhi + j;
          const float g = xe_lds[(size_t)(im * 16 + er) * H_DIM + h];
          ybuf[er * H_DIM + h] = acc[nt][j] * g;
        }
      }
    }
    asm volatile("s_wait_dscnt 0" ::: "memory");   // cross-lane LDS transpose hand-off

    // GEMM2: z(16x512) = y(16x256) @ w2[i][m]^T ; recombine into out
    #pragma unroll 1
    for (int nh = 0; nh < 2; ++nh) {
      v8f acc[16];
      #pragma unroll
      for (int t = 0; t < 16; ++t)
        #pragma unroll
        for (int j = 0; j < 8; ++j) acc[t][j] = 0.0f;

      gemm16<H_DIM / 32, 32>(ybuf, H_DIM,
                             wf + (size_t)(3072 + im * 256) * FRAG_HALVES, nh * 16, lane, acc);

      #pragma unroll
      for (int nt = 0; nt < 16; ++nt) {
        const int l = nh * 256 + nt * 16 + col;
        #pragma unroll
        for (int j = 0; j < 8; ++j) {
          const int er = rhi + j;
          float* dst = out + (size_t)(e0 + er) * 4096 + m * 1024 + p * 512 + l;
          const float v = sign * acc[nt][j];
          if (pipe == 0) *dst = v;                 // first pipeline defines the row
          else           *dst = *dst + v;         // same wave/lane: ordered store->load
        }
      }
    }
  }
}

extern "C" void kernel_launch(void* const* d_in, const int* in_sizes, int n_in,
                              void* d_out, int out_size, void* d_ws, size_t ws_size,
                              hipStream_t stream) {
  (void)in_sizes; (void)n_in; (void)ws_size; (void)out_size;
  const float* x    = (const float*)d_in[0];
  const float* xedg = (const float*)d_in[1];
  const float* Wd   = (const float*)d_in[2];
  const float* bd   = (const float*)d_in[3];
  const float* w1   = (const float*)d_in[4];
  const float* w2   = (const float*)d_in[5];
  float*  out = (float*)d_out;
  __bf16* wf  = (__bf16*)d_ws;            // needs 5120 * 2 KB = 10 MB scratch

  // one-time (per launch) weight split + swizzle; same stream -> ordered before main kernel
  prep_weights<<<dim3((NFRAG_TOTAL + 7) / 8), dim3(256), 0, stream>>>(Wd, w1, w2, wf);

  const size_t shmem = (size_t)(32768 + 8 * 16 * H_DIM) * sizeof(float);  // 256 KB (<=320 KB/WGP)
  (void)hipFuncSetAttribute((const void*)fused_layerblock,
                            hipFuncAttributeMaxDynamicSharedMemorySize, (int)shmem);
  fused_layerblock<<<dim3(E_TOT / 16), dim3(256), shmem, stream>>>(
      x, xedg, bd, wf, out);
}
